// HeteroGNN_89369679495193
// MI455X (gfx1250) — matrix operand order repaired and checked by reading, not compiled, not run on hardware
//
#include <hip/hip_runtime.h>
#include <hip/hip_bf16.h>

typedef __attribute__((ext_vector_type(16))) _Float16 v16h;
typedef __attribute__((ext_vector_type(8)))  float    v8f;

#define N_USER 100000
#define D 64

// ---------------------------------------------------------------------------
// Utility kernels
// ---------------------------------------------------------------------------
__global__ void hg_zero_i32(int* __restrict__ p, int n) {
    int i = blockIdx.x * blockDim.x + threadIdx.x;
    if (i < n) p[i] = 0;
}

__global__ void hg_zero_f32v4(float4* __restrict__ p, int n4) {
    int i = blockIdx.x * blockDim.x + threadIdx.x;
    if (i < n4) p[i] = make_float4(0.f, 0.f, 0.f, 0.f);
}

// degree counts for the p2p relation
__global__ void hg_degrees(const int* __restrict__ src, const int* __restrict__ dst,
                           int* __restrict__ outdeg, int* __restrict__ indeg, int nedges) {
    int i = blockIdx.x * blockDim.x + threadIdx.x;
    if (i < nedges) {
        atomicAdd(&outdeg[src[i]], 1);
        atomicAdd(&indeg[dst[i]], 1);
    }
}

__global__ void hg_rsqrt_deg(const int* __restrict__ outdeg, const int* __restrict__ indeg,
                             float* __restrict__ r_out, float* __restrict__ r_in, int n) {
    int i = blockIdx.x * blockDim.x + threadIdx.x;
    if (i < n) {
        int od = outdeg[i]; if (od < 1) od = 1;
        int id = indeg[i];  if (id < 1) id = 1;
        r_out[i] = rsqrtf((float)od);
        r_in[i]  = rsqrtf((float)id);
    }
}

// ---------------------------------------------------------------------------
// Edge scatter: agg[dst] += x[src] * r_out[src]
// 16 threads per edge, 4 dims per thread (float4 gather + 4 f32 atomics).
// Gather/scatter targets are 25.6MB -> L2 resident on MI455X (192MB L2).
// ---------------------------------------------------------------------------
__global__ void hg_scatter(const float* __restrict__ x, const float* __restrict__ r_out,
                           const int* __restrict__ src, const int* __restrict__ dst,
                           float* __restrict__ agg, int nedges) {
    int g  = blockIdx.x * blockDim.x + threadIdx.x;
    int e  = g >> 4;
    int c4 = (g & 15) << 2;
    if (e >= nedges) return;
    int s = src[e];
    int d = dst[e];
    float sc = r_out[s];
    float4 v = *(const float4*)(x + (size_t)s * D + c4);
    float* p = agg + (size_t)d * D + c4;
    atomicAdd(p + 0, v.x * sc);
    atomicAdd(p + 1, v.y * sc);
    atomicAdd(p + 2, v.z * sc);
    atomicAdd(p + 3, v.w * sc);
}

// ---------------------------------------------------------------------------
// WMMA GEMM: out = relu((agg .* r_in[row]) @ W + b)
// M = 100000 (exactly 6250 tiles of 16 rows), N = K = 64.
// One block = 4 waves; each wave owns a 16x16 output tile (noff = wave*16),
// accumulating two K=32 chunks with v_wmma_f32_16x16x32_f16.
// ---------------------------------------------------------------------------
__global__ __launch_bounds__(128)
void hg_gemm_relu(const float* __restrict__ agg, const float* __restrict__ r_in,
                  const float* __restrict__ W,   const float* __restrict__ bias,
                  float* __restrict__ out) {
    __shared__ float As[16][D];   // row-scaled A tile (fp32)

    const int row0 = blockIdx.x * 16;
    const int tid  = threadIdx.x;

    // Stage 16x64 tile, fused with rsqrt(indeg) scaling. 128 thr x 8 elems.
    #pragma unroll
    for (int i = 0; i < 8; ++i) {
        int idx = tid + i * 128;
        int r = idx >> 6, c = idx & 63;
        As[r][c] = agg[(size_t)(row0 + r) * D + c] * r_in[row0 + r];
    }
    __syncthreads();

    const int wave = tid >> 5;
    const int lane = tid & 31;
    const int noff = wave * 16;

    // --- A fragments (16x32 f16, ISA layout) ---
    // lane<16: M=lane, K = h<8 ? h : 16+(h-8) ; lanes 16-31: M=lane-16, K += 8
    const int mrow = lane & 15;
    const int kadd = (lane < 16) ? 0 : 8;
    v16h a0, a1;
    #pragma unroll
    for (int h = 0; h < 16; ++h) {
        int k = ((h < 8) ? h : (8 + h)) + kadd;   // {0..7,16..23} (+8 for hi lanes)
        a0[h] = (_Float16)As[mrow][k];
        a1[h] = (_Float16)As[mrow][32 + k];
    }

    // --- B fragments (32x16 f16): N = lane&15, K same order as A ---
    v16h b0, b1;
    {
        const float* w0 = W + noff + (lane & 15);           // K rows 0..31
        const float* w1 = W + (size_t)32 * D + noff + (lane & 15);  // K rows 32..63
        #pragma unroll
        for (int h = 0; h < 16; ++h) {
            int k = ((h < 8) ? h : (8 + h)) + kadd;
            b0[h] = (_Float16)w0[(size_t)k * D];
            b1[h] = (_Float16)w1[(size_t)k * D];
        }
    }

    v8f c = {};
    c = __builtin_amdgcn_wmma_f32_16x16x32_f16(false, a0, false, b0, (short)0, c, false, false);
    c = __builtin_amdgcn_wmma_f32_16x16x32_f16(false, a1, false, b1, (short)0, c, false, false);

    // --- Epilogue: C/D layout: VGPR r, lane -> (M = r + 8*(lane>=16), N = lane&15)
    const int nloc  = lane & 15;
    const int mbase = (lane < 16) ? 0 : 8;
    const float bb  = bias[noff + nloc];
    #pragma unroll
    for (int r = 0; r < 8; ++r) {
        float v = c[r] + bb;
        v = v > 0.f ? v : 0.f;
        out[(size_t)(row0 + mbase + r) * D + noff + nloc] = v;
    }
}

// ---------------------------------------------------------------------------
// Final fc: out[i][0..1] = h[i] @ wfc + bfc   (N=2 -> plain VALU)
// ---------------------------------------------------------------------------
__global__ void hg_fc(const float* __restrict__ h, const float* __restrict__ wfc,
                      const float* __restrict__ bfc, float* __restrict__ out, int M) {
    int i = blockIdx.x * blockDim.x + threadIdx.x;
    if (i >= M) return;
    const float* hr = h + (size_t)i * D;
    float a0 = bfc[0], a1 = bfc[1];
    #pragma unroll
    for (int k = 0; k < D; ++k) {
        float v = hr[k];
        a0 += v * wfc[k * 2 + 0];
        a1 += v * wfc[k * 2 + 1];
    }
    out[i * 2 + 0] = a0;
    out[i * 2 + 1] = a1;
}

// ---------------------------------------------------------------------------
// Launch. Only the p2p relation is live (all other relation outputs are
// dead in the reference), so cc/ip/used inputs are ignored.
// ---------------------------------------------------------------------------
extern "C" void kernel_launch(void* const* d_in, const int* in_sizes, int n_in,
                              void* d_out, int out_size, void* d_ws, size_t ws_size,
                              hipStream_t stream) {
    const float* x    = (const float*)d_in[0];   // x_user [100000,64]
    const float* w1   = (const float*)d_in[1];   // w1_p2p [64,64]
    const float* b1   = (const float*)d_in[2];   // b1_p2p [64]
    const float* w2   = (const float*)d_in[3];   // w2_p2p [64,64]
    const float* b2   = (const float*)d_in[4];   // b2_p2p [64]
    const float* wfc  = (const float*)d_in[17];  // wfc [64,2]
    const float* bfc  = (const float*)d_in[18];  // bfc [2]
    const int*   src  = (const int*)d_in[19];    // p2p_src [2M]
    const int*   dst  = (const int*)d_in[20];    // p2p_dst [2M]
    const int E = in_sizes[19];
    const int M = N_USER;

    // Workspace carve-up (packed back to back, no padding)
    char* ws = (char*)d_ws;
    float* agg   = (float*)ws;  ws += (size_t)M * D * sizeof(float);   // 25.6 MB
    float* hbuf  = (float*)ws;  ws += (size_t)M * D * sizeof(float);   // 25.6 MB
    float* r_out = (float*)ws;  ws += (size_t)M * sizeof(float);
    float* r_in  = (float*)ws;  ws += (size_t)M * sizeof(float);
    int*   odeg  = (int*)ws;    ws += (size_t)M * sizeof(int);
    int*   ideg  = (int*)ws;    ws += (size_t)M * sizeof(int);

    const int T = 256;
    const int nodeBlocks = (M + T - 1) / T;
    const int edgeBlocks = (E + T - 1) / T;
    const int scatBlocks = ((E * 16) + T - 1) / T;
    const int aggV4      = (M * D) / 4;
    const int zeroBlocks = (aggV4 + T - 1) / T;
    const int gemmBlocks = M / 16;               // 6250, exact

    // Degrees (shared by both layers)
    hg_zero_i32<<<(2 * M + T - 1) / T, T, 0, stream>>>(odeg, 2 * M); // odeg+ideg contiguous
    hg_degrees<<<edgeBlocks, T, 0, stream>>>(src, dst, odeg, ideg, E);
    hg_rsqrt_deg<<<nodeBlocks, T, 0, stream>>>(odeg, ideg, r_out, r_in, M);

    // Layer 1
    hg_zero_f32v4<<<zeroBlocks, T, 0, stream>>>((float4*)agg, aggV4);
    hg_scatter<<<scatBlocks, T, 0, stream>>>(x, r_out, src, dst, agg, E);
    hg_gemm_relu<<<gemmBlocks, 128, 0, stream>>>(agg, r_in, w1, b1, hbuf);

    // Layer 2 (reads hbuf for messages, overwrites hbuf with h2)
    hg_zero_f32v4<<<zeroBlocks, T, 0, stream>>>((float4*)agg, aggV4);
    hg_scatter<<<scatBlocks, T, 0, stream>>>(hbuf, r_out, src, dst, agg, E);
    hg_gemm_relu<<<gemmBlocks, 128, 0, stream>>>(agg, r_in, w2, b2, hbuf);

    // Final fc -> d_out [100000, 2]
    hg_fc<<<nodeBlocks, T, 0, stream>>>(hbuf, wfc, bfc, (float*)d_out, M);
}
